// LearningWithinSingleSpinConfiguration_55705725829286
// MI455X (gfx1250) — matrix operand order repaired and checked by reading, not compiled, not run on hardware
//
#include <hip/hip_runtime.h>
#include <hip/hip_bf16.h>
#include <stdint.h>

// ---------------------------------------------------------------------------
// MI455X (gfx1250) fused GNN message-passing pipeline, round 2.
// v_wmma_f32_16x16x32_bf16 with M=64 register blocking: each packed B
// fragment (one 32B load/lane) is reused by 4 WMMAs via 4 accumulators.
// Hidden activations never leave LDS (320KB/WGP budget, ~100KB/block).
// ---------------------------------------------------------------------------

typedef __attribute__((ext_vector_type(16))) __bf16 v16bf;
typedef __attribute__((ext_vector_type(8)))  __bf16 v8bf;
typedef __attribute__((ext_vector_type(8)))  float  v8f;

struct MlpW {
    const __bf16* w0; const __bf16* w1; const __bf16* w2;
    const float*  b0; const float*  b1; const float*  b2;
};

// ---------------------------------------------------------------------------
// Weight packing: W[Ksrc,Nsrc] (row-major f32, as in x @ W) -> bf16 tiles in
// the WMMA B-lane layout. Tile (j = N/16, t = K/32): 32 lanes x 16 halves,
// so each lane's B fragment in the GEMM is one contiguous 32B load.
// ---------------------------------------------------------------------------
__global__ void k_pack(const float* __restrict__ W, __bf16* __restrict__ dst,
                       int Ksrc, int Nsrc, int Kpad, int Npad) {
    int KT = Kpad >> 5, NT = Npad >> 4;
    int total = NT * KT * 32;
    int i = blockIdx.x * blockDim.x + threadIdx.x;
    if (i >= total) return;
    int lane = i & 31;
    int tile = i >> 5;
    int t = tile % KT;
    int j = tile / KT;
    int n  = j * 16 + (lane & 15);
    int k0 = t * 32 + ((lane < 16) ? 0 : 8);
    v16bf pk;
#pragma unroll
    for (int h = 0; h < 16; ++h) {
        int k = k0 + ((h < 8) ? h : h + 8);
        float v = (k < Ksrc && n < Nsrc) ? W[(size_t)k * Nsrc + n] : 0.f;
        pk[h] = (__bf16)v;
    }
    *(v16bf*)(dst + (size_t)i * 16) = pk;   // 32B-aligned contiguous store
}

// Load one A fragment (ISA 16-bit A layout) from row-major bf16 LDS.
__device__ __forceinline__
v16bf load_a(const __bf16* arow) {           // arow already lane-adjusted
    v8bf lo = *(const v8bf*)(arow);          // K kb..kb+7
    v8bf hi = *(const v8bf*)(arow + 16);     // K 16+kb..16+kb+7
    v16bf a;
#pragma unroll
    for (int i = 0; i < 8; ++i) { a[i] = lo[i]; a[i + 8] = hi[i]; }
    return a;
}

// ---------------------------------------------------------------------------
// Single 16x16 output tile (used by the small final layers).
// ---------------------------------------------------------------------------
__device__ __forceinline__
v8f gemm16(const __bf16* sA, int ldk, int KT,
           const __bf16* wp, int jt, float binit) {
    int lane = threadIdx.x & 31;
    int r  = lane & 15;
    int kb = (lane < 16) ? 0 : 8;
    v8f c;
#pragma unroll
    for (int i = 0; i < 8; ++i) c[i] = binit;
    const __bf16* arow = sA + r * ldk + kb;
    const __bf16* wt   = wp + (size_t)jt * KT * 512 + lane * 16;
    for (int t = 0; t < KT; ++t) {
        v16bf a = load_a(arow + t * 32);
        v16bf b = *(const v16bf*)(wt + (size_t)t * 512);
        c = __builtin_amdgcn_wmma_f32_16x16x32_bf16(
                false, a, false, b, (short)0, c, false, false);
    }
    return c;
}

// ---------------------------------------------------------------------------
// Hidden layer over MT*16 rows with M-register blocking:
//   sA[MT*16 x K] -> sOut[MT*16 x N]; each B fragment feeds MT WMMAs.
// ---------------------------------------------------------------------------
template<int MT>
__device__ __forceinline__
void layerM(const __bf16* sA, int K, __bf16* sOut, int N,
            const __bf16* wp, const float* bias, bool relu) {
    int wave = threadIdx.x >> 5;
    int nw   = blockDim.x >> 5;
    int lane = threadIdx.x & 31;
    int KT = K >> 5, NT = N >> 4;
    int r  = lane & 15;
    int kb = (lane < 16) ? 0 : 8;
    for (int j = wave; j < NT; j += nw) {
        if (j + nw < NT)                     // gfx1250 global_prefetch_b8
            __builtin_prefetch(wp + (size_t)(j + nw) * KT * 512, 0, 0);
        float bv = bias[j * 16 + (lane & 15)];
        v8f acc[MT];
#pragma unroll
        for (int mt = 0; mt < MT; ++mt)
#pragma unroll
            for (int i = 0; i < 8; ++i) acc[mt][i] = bv;
        const __bf16* wt = wp + (size_t)j * KT * 512 + lane * 16;
        for (int t = 0; t < KT; ++t) {
            v16bf b = *(const v16bf*)(wt + (size_t)t * 512);
#pragma unroll
            for (int mt = 0; mt < MT; ++mt) {
                v16bf a = load_a(sA + (size_t)(mt * 16 + r) * K + kb + t * 32);
                acc[mt] = __builtin_amdgcn_wmma_f32_16x16x32_bf16(
                              false, a, false, b, (short)0, acc[mt], false, false);
            }
        }
        int n  = j * 16 + (lane & 15);
        int mb = (lane < 16) ? 0 : 8;
#pragma unroll
        for (int mt = 0; mt < MT; ++mt)
#pragma unroll
            for (int e = 0; e < 8; ++e) {
                float v = acc[mt][e];
                if (relu) v = fmaxf(v, 0.f);
                sOut[(size_t)(mt * 16 + mb + e) * N + n] = (__bf16)v;
            }
    }
    __syncthreads();
}

// ---------------------------------------------------------------------------
// Fused edge kernel: 64 edges per block (4 row-tiles).
//  stage [x_j | x_i | ea] (160 bf16 cols) -> chi(160,320,320,32)
//  -> overwrite ea slot with new_ea -> phi(160,320,320,80)
//  -> atomic scatter of messages onto msg_sum[dst].
// ---------------------------------------------------------------------------
__global__ __launch_bounds__(256)
void k_edge(const float* __restrict__ x, const long long* __restrict__ ei,
            const float* ea_f32, const __bf16* ea_bf16,
            MlpW chi, MlpW phi,
            float* eaout_f32, __bf16* eaout_bf16,
            float* __restrict__ msg_sum, int E) {
    __shared__ __attribute__((aligned(32))) __bf16 sIn[64 * 160];   // 20 KB
    __shared__ __attribute__((aligned(32))) __bf16 sH1[64 * 320];   // 40 KB
    __shared__ __attribute__((aligned(32))) __bf16 sH2[64 * 320];   // 40 KB
    __shared__ int sDst[64];

    int tid = threadIdx.x;
    int e0  = blockIdx.x * 64;

    for (int idx = tid; idx < 64 * 160; idx += blockDim.x) {
        int r = idx / 160, c = idx - r * 160;
        int e = e0 + r;
        float v = 0.f;
        if (e < E) {
            if (c < 64) {
                v = x[(size_t)ei[e] * 64 + c];
            } else if (c < 128) {
                v = x[(size_t)ei[(size_t)E + e] * 64 + (c - 64)];
            } else {
                v = ea_bf16 ? (float)ea_bf16[(size_t)e * 32 + (c - 128)]
                            : ea_f32[(size_t)e * 32 + (c - 128)];
            }
        }
        sIn[idx] = (__bf16)v;
    }
    if (tid < 64) {
        int e = e0 + tid;
        sDst[tid] = (e < E) ? (int)ei[(size_t)E + e] : 0;
    }
    __syncthreads();

    // ---- chi: edge-attr update MLP ----
    layerM<4>(sIn, 160, sH1, 320, chi.w0, chi.b0, true);
    layerM<4>(sH1, 320, sH2, 320, chi.w1, chi.b1, true);
    {
        int wave = tid >> 5, nw = blockDim.x >> 5, lane = tid & 31;
        for (int tidx = wave; tidx < 4 * 2; tidx += nw) {   // 4 mtiles x 2 ntiles
            int mt = tidx >> 1, j = tidx & 1;
            float bv = chi.b2[j * 16 + (lane & 15)];
            v8f c = gemm16(sH2 + (size_t)mt * 16 * 320, 320, 10, chi.w2, j, bv);
            int n  = j * 16 + (lane & 15);
            int mb = (lane < 16) ? 0 : 8;
#pragma unroll
            for (int el = 0; el < 8; ++el) {
                int m = mt * 16 + mb + el, e = e0 + m;
                float v = c[el];
                sIn[m * 160 + 128 + n] = (__bf16)v;         // becomes phi input
                if (e < E) {
                    if (eaout_f32) eaout_f32[(size_t)e * 32 + n] = v;
                    else           eaout_bf16[(size_t)e * 32 + n] = (__bf16)v;
                }
            }
        }
    }
    __syncthreads();

    // ---- phi: message MLP ----
    layerM<4>(sIn, 160, sH1, 320, phi.w0, phi.b0, true);
    layerM<4>(sH1, 320, sH2, 320, phi.w1, phi.b1, true);
    {
        int wave = tid >> 5, nw = blockDim.x >> 5, lane = tid & 31;
        for (int tidx = wave; tidx < 4 * 5; tidx += nw) {   // 4 mtiles x 5 ntiles
            int mt = tidx / 5, j = tidx - mt * 5;
            float bv = phi.b2[j * 16 + (lane & 15)];
            v8f c = gemm16(sH2 + (size_t)mt * 16 * 320, 320, 10, phi.w2, j, bv);
            int n  = j * 16 + (lane & 15);
            int mb = (lane < 16) ? 0 : 8;
#pragma unroll
            for (int el = 0; el < 8; ++el) {
                int m = mt * 16 + mb + el, e = e0 + m;
                if (e < E)
                    atomicAdd(&msg_sum[(size_t)sDst[m] * 80 + n], c[el]);
            }
        }
    }
}

// ---------------------------------------------------------------------------
// Fused node kernel: 64 nodes per block. input = [mean_msg(80) | x(64) | 0pad]
// gamma(160(144),288,288,64) -> xout (f32).
// ---------------------------------------------------------------------------
__global__ __launch_bounds__(256)
void k_node(const float* __restrict__ xin, const float* __restrict__ msg_sum,
            const float* __restrict__ cnt, MlpW gam,
            float* __restrict__ xout, int NN) {
    __shared__ __attribute__((aligned(32))) __bf16 sIn[64 * 160];   // 20 KB
    __shared__ __attribute__((aligned(32))) __bf16 sH1[64 * 288];   // 36 KB
    __shared__ __attribute__((aligned(32))) __bf16 sH2[64 * 288];   // 36 KB

    int tid = threadIdx.x;
    int n0  = blockIdx.x * 64;

    for (int idx = tid; idx < 64 * 160; idx += blockDim.x) {
        int r = idx / 160, c = idx - r * 160;
        int nd = n0 + r;
        float v = 0.f;
        if (nd < NN) {
            if (c < 80)       v = msg_sum[(size_t)nd * 80 + c] / fmaxf(cnt[nd], 1.f);
            else if (c < 144) v = xin[(size_t)nd * 64 + (c - 80)];
        }
        sIn[idx] = (__bf16)v;
    }
    __syncthreads();

    layerM<4>(sIn, 160, sH1, 288, gam.w0, gam.b0, true);
    layerM<4>(sH1, 288, sH2, 288, gam.w1, gam.b1, true);

    int wave = tid >> 5, nw = blockDim.x >> 5, lane = tid & 31;
    for (int tidx = wave; tidx < 4 * 4; tidx += nw) {       // 4 mtiles x 4 ntiles
        int mt = tidx >> 2, j = tidx & 3;
        float bv = gam.b2[j * 16 + (lane & 15)];
        v8f c = gemm16(sH2 + (size_t)mt * 16 * 288, 288, 9, gam.w2, j, bv);
        int n  = j * 16 + (lane & 15);
        int mb = (lane < 16) ? 0 : 8;
#pragma unroll
        for (int el = 0; el < 8; ++el) {
            int nd = n0 + mt * 16 + mb + el;
            if (nd < NN) xout[(size_t)nd * 64 + n] = c[el];
        }
    }
}

// ---------------------------------------------------------------------------
// Graph pooling: atomic segment-sum + per-segment element count.
// ---------------------------------------------------------------------------
__global__ void k_pool(const float* __restrict__ vals, const long long* __restrict__ ids,
                       float* __restrict__ sum, float* __restrict__ cnt,
                       long rows, int cols) {
    long i = (long)blockIdx.x * blockDim.x + threadIdx.x;
    if (i >= rows * cols) return;
    long r = i / cols;
    int  c = (int)(i - r * cols);
    int  g = (int)ids[r];
    atomicAdd(&sum[(size_t)g * cols + c], vals[i]);
    if (c == 0) atomicAdd(&cnt[g], 1.f);
}

__global__ void k_count_dst(const long long* __restrict__ ei,
                            float* __restrict__ cnt, int E) {
    int e = blockIdx.x * blockDim.x + threadIdx.x;
    if (e < E) atomicAdd(&cnt[(int)ei[(size_t)E + e]], 1.f);
}

__global__ void k_zero(float* __restrict__ p, long n) {
    long i = (long)blockIdx.x * blockDim.x + threadIdx.x;
    if (i < n) p[i] = 0.f;
}

// ---------------------------------------------------------------------------
// alpha head: 64 graphs, input [pooled_x(64) | pooled_ea(32)] = 96 cols.
// 96 -> 192 -> 192 -> 16(pad of 5). Single block, M=64 register blocked.
// ---------------------------------------------------------------------------
__global__ __launch_bounds__(256)
void k_alpha(const float* __restrict__ nsum, const float* __restrict__ ncnt,
             const float* __restrict__ esum, const float* __restrict__ ecnt,
             MlpW al, float* __restrict__ coeff) {
    __shared__ __attribute__((aligned(32))) __bf16 sIn[64 * 96];    // 12 KB
    __shared__ __attribute__((aligned(32))) __bf16 sH1[64 * 192];   // 24 KB
    __shared__ __attribute__((aligned(32))) __bf16 sH2[64 * 192];   // 24 KB

    int tid = threadIdx.x;
    for (int idx = tid; idx < 64 * 96; idx += blockDim.x) {
        int g = idx / 96, c = idx - g * 96;
        float v = (c < 64) ? nsum[g * 64 + c] / fmaxf(ncnt[g], 1.f)
                           : esum[g * 32 + (c - 64)] / fmaxf(ecnt[g], 1.f);
        sIn[idx] = (__bf16)v;
    }
    __syncthreads();

    layerM<4>(sIn, 96, sH1, 192, al.w0, al.b0, true);
    layerM<4>(sH1, 192, sH2, 192, al.w1, al.b1, true);

    int wave = tid >> 5, lane = tid & 31;
    if (wave < 4) {                                   // 4 tiles of [16 x 16]
        int nn = lane & 15;
        float bv = (nn < 5) ? al.b2[nn] : 0.f;
        v8f c = gemm16(sH2 + (size_t)wave * 16 * 192, 192, 6, al.w2, 0, bv);
        int mb = (lane < 16) ? 0 : 8;
#pragma unroll
        for (int el = 0; el < 8; ++el) {
            int g = wave * 16 + mb + el;
            if (nn < 5) coeff[g * 5 + nn] = c[el];
        }
    }
}

// ---------------------------------------------------------------------------
// Host orchestration
// ---------------------------------------------------------------------------
extern "C" void kernel_launch(void* const* d_in, const int* in_sizes, int n_in,
                              void* d_out, int out_size, void* d_ws, size_t ws_size,
                              hipStream_t stream) {
    (void)n_in; (void)out_size; (void)ws_size;
    const float*     x  = (const float*)d_in[0];
    const long long* ei = (const long long*)d_in[1];   // int64 [2,E]
    const float*     ea = (const float*)d_in[2];
    const long long* nb = (const long long*)d_in[3];
    const long long* eb = (const long long*)d_in[4];
    const int NN = in_sizes[0] / 64;
    const int E  = in_sizes[1] / 2;

    // params flattened in dict-insertion order:
    //  layer1{chi,phi,gamma} layer2{chi,phi,gamma} alpha, each as (W,b)x3
    //  W_i at d_in[5+2i], b_i at d_in[6+2i].
    static const int WD[21][4] = {   // {Ksrc,Nsrc,Kpad,Npad}
        {160,320,160,320},{320,320,320,320},{320, 32,320, 32},   // L1 chi
        {160,320,160,320},{320,320,320,320},{320, 80,320, 80},   // L1 phi
        {144,288,160,288},{288,288,288,288},{288, 64,288, 64},   // L1 gamma
        {160,320,160,320},{320,320,320,320},{320, 32,320, 32},   // L2 chi
        {160,320,160,320},{320,320,320,320},{320, 80,320, 80},   // L2 phi
        {144,288,160,288},{288,288,288,288},{288, 64,288, 64},   // L2 gamma
        { 96,192, 96,192},{192,192,192,192},{192,  5,192, 16},   // alpha
    };

    char* ws = (char*)d_ws;
    size_t cur = 0;
    auto carve = [&](size_t bytes) -> char* {
        char* p = ws + cur;
        cur = (cur + bytes + 255) & ~(size_t)255;
        return p;
    };

    __bf16* pw[21];
    for (int i = 0; i < 21; ++i)
        pw[i] = (__bf16*)carve((size_t)WD[i][2] * WD[i][3] * 2);
    float*  msg_sum = (float*)carve((size_t)NN * 80 * 4);
    float*  cnt_dst = (float*)carve((size_t)NN * 4);
    __bf16* ea1     = (__bf16*)carve((size_t)E * 32 * 2);
    float*  x1      = (float*)carve((size_t)NN * 64 * 4);
    float*  psum_n  = (float*)carve(64 * 64 * 4);
    float*  psum_e  = (float*)carve(64 * 32 * 4);
    float*  pcnt_n  = (float*)carve(64 * 4);
    float*  pcnt_e  = (float*)carve(64 * 4);

    // 1) pack all weights to bf16 WMMA tile layout (~2MB; lives in L2)
    for (int i = 0; i < 21; ++i) {
        int lanes = (WD[i][2] >> 5) * (WD[i][3] >> 4) * 32;
        k_pack<<<(lanes + 255) / 256, 256, 0, stream>>>(
            (const float*)d_in[5 + 2 * i], pw[i],
            WD[i][0], WD[i][1], WD[i][2], WD[i][3]);
    }

    auto mlp = [&](int base) -> MlpW {
        MlpW m;
        m.w0 = pw[base];     m.w1 = pw[base + 1]; m.w2 = pw[base + 2];
        m.b0 = (const float*)d_in[6 + 2 * base];
        m.b1 = (const float*)d_in[6 + 2 * (base + 1)];
        m.b2 = (const float*)d_in[6 + 2 * (base + 2)];
        return m;
    };
    MlpW chi1 = mlp(0),  phi1 = mlp(3),  gam1 = mlp(6);
    MlpW chi2 = mlp(9),  phi2 = mlp(12), gam2 = mlp(15);
    MlpW alph = mlp(18);

    float* ea2   = (float*)d_out;
    float* x2    = ea2 + (size_t)E * 32;
    float* coeff = x2 + (size_t)NN * 64;

    // 2) in-degree counts (identical for both layers)
    k_zero<<<(NN + 255) / 256, 256, 0, stream>>>(cnt_dst, NN);
    k_count_dst<<<(E + 255) / 256, 256, 0, stream>>>(ei, cnt_dst, E);

    // 3) layer 1
    k_zero<<<(int)(((long)NN * 80 + 255) / 256), 256, 0, stream>>>(msg_sum, (long)NN * 80);
    k_edge<<<(E + 63) / 64, 256, 0, stream>>>(x, ei, ea, (const __bf16*)nullptr,
                                              chi1, phi1,
                                              (float*)nullptr, ea1, msg_sum, E);
    k_node<<<(NN + 63) / 64, 256, 0, stream>>>(x, msg_sum, cnt_dst, gam1, x1, NN);

    // 4) layer 2
    k_zero<<<(int)(((long)NN * 80 + 255) / 256), 256, 0, stream>>>(msg_sum, (long)NN * 80);
    k_edge<<<(E + 63) / 64, 256, 0, stream>>>(x1, ei, (const float*)nullptr, ea1,
                                              chi2, phi2,
                                              ea2, (__bf16*)nullptr, msg_sum, E);
    k_node<<<(NN + 63) / 64, 256, 0, stream>>>(x1, msg_sum, cnt_dst, gam2, x2, NN);

    // 5) pooling + alpha head
    k_zero<<<17, 256, 0, stream>>>(psum_n, 64 * 64);
    k_zero<<<9, 256, 0, stream>>>(psum_e, 64 * 32);
    k_zero<<<1, 256, 0, stream>>>(pcnt_n, 64);
    k_zero<<<1, 256, 0, stream>>>(pcnt_e, 64);
    k_pool<<<(int)(((long)NN * 64 + 255) / 256), 256, 0, stream>>>(x2, nb, psum_n, pcnt_n, NN, 64);
    k_pool<<<(int)(((long)E * 32 + 255) / 256), 256, 0, stream>>>(ea2, eb, psum_e, pcnt_e, E, 32);
    k_alpha<<<1, 256, 0, stream>>>(psum_n, pcnt_n, psum_e, pcnt_e, alph, coeff);
}
